// RecurrentBDH_48687749267456
// MI455X (gfx1250) — matrix operand order repaired and checked by reading, not compiled, not run on hardware
//
#include <hip/hip_runtime.h>
#include <hip/hip_bf16.h>
#include <math.h>

// ---------------------------------------------------------------------------
// RecurrentBDH forward for MI455X (gfx1250, wave32, WMMA).
// All GEMMs run through v_wmma_f32_16x16x32_bf16 (bf16 in, f32 accumulate).
// Block tile 128x64, 8 waves, wave tile 32x32 => 4 WMMA / wave / K-step.
// Async global->LDS staging (GLOBAL_LOAD_ASYNC_TO_LDS_B128, ASYNCcnt).
// ---------------------------------------------------------------------------

typedef __bf16 bf16;
typedef bf16  v16bf __attribute__((ext_vector_type(16)));
typedef bf16  v8bf  __attribute__((ext_vector_type(8)));
typedef float v8f   __attribute__((ext_vector_type(8)));
typedef unsigned int u32;

#define T_SEQ 1024
#define D_DIM 512
#define NH    8
#define N_DIM 2048
#define V_DIM 32000
#define HN    (NH * N_DIM)          // 16384
#define HND   (NH * N_DIM * D_DIM)  // 8388608
#define NLAYER 4

// ---- CDNA5 async global->LDS path (guarded; fallback is manual) ----
#if defined(__has_builtin)
#if __has_builtin(__builtin_amdgcn_global_load_async_to_lds_b128) && \
    __has_builtin(__builtin_amdgcn_s_wait_asynccnt)
#define USE_ASYNC_LDS 1
#endif
#endif
#ifndef USE_ASYNC_LDS
#define USE_ASYNC_LDS 0
#endif

#if USE_ASYNC_LDS
typedef int v4i __attribute__((vector_size(16)));
__device__ __forceinline__ void cp_async16(const bf16* g, bf16* l) {
  // 16 bytes per lane, global -> LDS, tracked by ASYNCcnt.
  // Builtin signature (from probe): (v4i AS1*, v4i AS3*, imm offset, imm cpol)
  __builtin_amdgcn_global_load_async_to_lds_b128(
      (__attribute__((address_space(1))) v4i*)(unsigned long long)g,
      (__attribute__((address_space(3))) v4i*)(unsigned int)(unsigned long long)l,
      0, 0);
}
#endif

// ---------------- elementwise fp32 -> bf16 ----------------
__global__ void k_f32_to_bf16(const float* __restrict__ in, bf16* __restrict__ out, long n) {
  long i = (long)blockIdx.x * blockDim.x + threadIdx.x;
  long stride = (long)gridDim.x * blockDim.x;
  for (; i < n; i += stride) out[i] = (bf16)in[i];
}

// ---------------- block-wide mean/rstd over 512 values (2 per thread) ----------------
__device__ inline void block_meanvar(float sum, float sumsq, float* s1, float* s2,
                                     float& mean, float& rstd) {
  int tid = threadIdx.x;
  s1[tid] = sum; s2[tid] = sumsq;
  __syncthreads();
  for (int off = 128; off > 0; off >>= 1) {
    if (tid < off) { s1[tid] += s1[tid + off]; s2[tid] += s2[tid + off]; }
    __syncthreads();
  }
  mean = s1[0] * (1.0f / D_DIM);
  float var = s2[0] * (1.0f / D_DIM) - mean * mean;
  rstd = rsqrtf(var + 1e-5f);
  __syncthreads();  // allow shared reuse
}

// ---------------- embedding gather + LayerNorm ----------------
__global__ void k_embed_ln(const int* __restrict__ idx, const float* __restrict__ embed_w,
                           float* __restrict__ x_f, bf16* __restrict__ x_b) {
  __shared__ float s1[256], s2[256];
  int t = blockIdx.x, tid = threadIdx.x;
  const float* row = embed_w + (long)idx[t] * D_DIM;
  float v0 = row[tid], v1 = row[tid + 256];
  float mean, rstd;
  block_meanvar(v0 + v1, v0 * v0 + v1 * v1, s1, s2, mean, rstd);
  float o0 = (v0 - mean) * rstd, o1 = (v1 - mean) * rstd;
  long base = (long)t * D_DIM;
  x_f[base + tid] = o0;  x_f[base + tid + 256] = o1;
  x_b[base + tid] = (bf16)o0; x_b[base + tid + 256] = (bf16)o1;
}

// ---------------- row LayerNorm (f32 in -> bf16 out), C = 512 ----------------
__global__ void k_ln_rows(const float* __restrict__ in, bf16* __restrict__ out) {
  __shared__ float s1[256], s2[256];
  long r = blockIdx.x; int tid = threadIdx.x;
  const float* p = in + r * D_DIM;
  float v0 = p[tid], v1 = p[tid + 256];
  float mean, rstd;
  block_meanvar(v0 + v1, v0 * v0 + v1 * v1, s1, s2, mean, rstd);
  out[r * D_DIM + tid]       = (bf16)((v0 - mean) * rstd);
  out[r * D_DIM + tid + 256] = (bf16)((v1 - mean) * rstd);
}

// ---------------- y = ln(yMLP); x = ln(x + y) ----------------
__global__ void k_ln_residual(const float* __restrict__ yMLP,
                              float* __restrict__ x_f, bf16* __restrict__ x_b) {
  __shared__ float s1[256], s2[256];
  long t = blockIdx.x; int tid = threadIdx.x;
  const float* yr = yMLP + t * D_DIM;
  float a0 = yr[tid], a1 = yr[tid + 256];
  float m1, r1;
  block_meanvar(a0 + a1, a0 * a0 + a1 * a1, s1, s2, m1, r1);
  float y0 = (a0 - m1) * r1, y1 = (a1 - m1) * r1;
  float z0 = x_f[t * D_DIM + tid] + y0, z1 = x_f[t * D_DIM + tid + 256] + y1;
  float m2, r2;
  block_meanvar(z0 + z1, z0 * z0 + z1 * z1, s1, s2, m2, r2);
  float o0 = (z0 - m2) * r2, o1 = (z1 - m2) * r2;
  x_f[t * D_DIM + tid] = o0;        x_f[t * D_DIM + tid + 256] = o1;
  x_b[t * D_DIM + tid] = (bf16)o0;  x_b[t * D_DIM + tid + 256] = (bf16)o1;
}

// ---------------- column mean over R rows (bf16 input, per head) ----------------
__global__ void k_colmean_bf16(const bf16* __restrict__ in, float* __restrict__ out,
                               int R, int C, long strideH) {
  int h = blockIdx.y;
  int col = blockIdx.x * blockDim.x + threadIdx.x;
  const bf16* p = in + (long)h * strideH + col;
  float s = 0.f;
  for (int r = 0; r < R; ++r) s += (float)p[(long)r * C];
  out[(long)h * C + col] = s * (1.0f / 1024.0f);
}

__global__ void k_colmean_f32(const float* __restrict__ in, float* __restrict__ out,
                              int R, int C) {
  int col = blockIdx.x * blockDim.x + threadIdx.x;
  float s = 0.f;
  for (int r = 0; r < R; ++r) s += in[(long)r * C + col];
  out[col] = s * (1.0f / 1024.0f);
}

// ---------------- rho finalize: upd = 0.25 * sum_l xm_l (x) vm_l ----------------
__global__ void k_rho_final(const float* __restrict__ xm, const float* __restrict__ vm,
                            const float* __restrict__ prev_rho,
                            float* __restrict__ out_new, float* __restrict__ out_upd) {
  long i = (long)blockIdx.x * 256 + threadIdx.x;   // i = (h*N + n)*D + d
  int d = (int)(i & (D_DIM - 1));
  long hn = i >> 9;                                // h*N + n
  float upd = 0.f;
  #pragma unroll
  for (int l = 0; l < NLAYER; ++l)
    upd += xm[(long)l * HN + hn] * vm[l * D_DIM + d];
  upd *= 0.25f;
  float pr = __builtin_nontemporal_load(&prev_rho[i]);
  __builtin_nontemporal_store(upd, &out_upd[i]);
  __builtin_nontemporal_store(0.99f * pr + upd, &out_new[i]);
}

// ---------------------------------------------------------------------------
// Tiled bf16 WMMA GEMM: C[M x Nout] = A[M x K] * B (B row-major [K x Nout],
// or TRANSB: B element [k][n] = Bmem[n][k]).  256 thr = 8 waves (4x2 grid),
// block tile 128x64, wave tile 32x32 => 4x v_wmma_f32_16x16x32_bf16 / K-step.
// Epilogues: 0 = f32 store; 1 = relu -> bf16 x_sparse + rope -> bf16 QR;
//            2 = strict-causal mask -> bf16; 3 = relu * aux -> bf16 (scatter);
//            4 = f32 non-temporal store (write-once logits).
// ---------------------------------------------------------------------------
template<int EPI, bool TRANSB>
__global__ __launch_bounds__(256)
void k_gemm(const bf16* __restrict__ A, const bf16* __restrict__ B, void* __restrict__ Cv,
            const bf16* __restrict__ aux, bf16* __restrict__ out2,
            int K, int lda, int ldb, int ldc, int ldaux,
            long sA, long sB, long sC, long sAux,
            const int* __restrict__ pos_off) {
  __shared__ __attribute__((aligned(16))) bf16 As[128 * 32];  // [row][k]  row-major
  __shared__ __attribute__((aligned(16))) bf16 Bs[64 * 32];   // [n][k]    (transposed)

  const int hz  = blockIdx.z;
  A += (long)hz * sA;
  B += (long)hz * sB;
  const int m0 = blockIdx.y * 128, n0 = blockIdx.x * 64;
  const int tid = threadIdx.x, lane = tid & 31, wave = tid >> 5;
  const int wm = wave & 3, wn = wave >> 2;      // wave tile: rows wm*32, cols wn*32
  const int hh = lane >> 4, rl = lane & 15;     // fragment lane decomposition

  v8f acc[2][2] = { { {0,0,0,0,0,0,0,0}, {0,0,0,0,0,0,0,0} },
                    { {0,0,0,0,0,0,0,0}, {0,0,0,0,0,0,0,0} } };

  for (int k0 = 0; k0 < K; k0 += 32) {
    // ---- stage A (128x32) and B^T (64x32) into LDS ----
#if USE_ASYNC_LDS
    // A: 512 x 16B chunks, 2 per thread (async, no VGPR round-trip)
    #pragma unroll
    for (int i = 0; i < 2; ++i) {
      int c = tid + 256 * i;                    // 0..511
      int r = c >> 2, q = c & 3;
      cp_async16(&A[(long)(m0 + r) * lda + k0 + q * 8], &As[r * 32 + q * 8]);
    }
    if (TRANSB) {                               // rows of Bmem are K-contiguous
      int n = tid >> 2, q = tid & 3;            // 256 x 16B chunks, 1 per thread
      cp_async16(&B[(long)(n0 + n) * ldb + k0 + q * 8], &Bs[n * 32 + q * 8]);
    } else {
      #pragma unroll
      for (int i = 0; i < 4; ++i) {             // coalesced read, transposed write
        int e = tid + 256 * i;
        int kk = e >> 5, nd = e & 31;
        u32 w = *(const u32*)&B[(long)(k0 + kk) * ldb + n0 + 2 * nd];
        Bs[(2 * nd)     * 32 + kk] = __builtin_bit_cast(bf16, (unsigned short)(w & 0xffffu));
        Bs[(2 * nd + 1) * 32 + kk] = __builtin_bit_cast(bf16, (unsigned short)(w >> 16));
      }
    }
    __builtin_amdgcn_s_wait_asynccnt(0);
#else
    #pragma unroll
    for (int i = 0; i < 8; ++i) {               // A: 2048 dwords, 8 per thread
      int e = tid + 256 * i;
      int r = e >> 4, kd = e & 15;
      *(u32*)&As[r * 32 + 2 * kd] =
          *(const u32*)&A[(long)(m0 + r) * lda + k0 + 2 * kd];
    }
    #pragma unroll
    for (int i = 0; i < 4; ++i) {
      int e = tid + 256 * i;                    // 0..1023
      if (TRANSB) {
        int n = e >> 4, kd = e & 15;
        *(u32*)&Bs[n * 32 + 2 * kd] =
            *(const u32*)&B[(long)(n0 + n) * ldb + k0 + 2 * kd];
      } else {
        int kk = e >> 5, nd = e & 31;
        u32 w = *(const u32*)&B[(long)(k0 + kk) * ldb + n0 + 2 * nd];
        Bs[(2 * nd)     * 32 + kk] = __builtin_bit_cast(bf16, (unsigned short)(w & 0xffffu));
        Bs[(2 * nd + 1) * 32 + kk] = __builtin_bit_cast(bf16, (unsigned short)(w >> 16));
      }
    }
#endif
    __syncthreads();

    // ---- A fragments: K(e) = 16*(e>>3) + 8*hh + (e&7) -> two contiguous v8bf ----
    v16bf afrag[2];
    #pragma unroll
    for (int mt = 0; mt < 2; ++mt) {
      const int ar = (wm * 32 + mt * 16 + rl) * 32;
      v8bf alo = *(const v8bf*)&As[ar + 8 * hh];
      v8bf ahi = *(const v8bf*)&As[ar + 16 + 8 * hh];
      afrag[mt] = __builtin_shufflevector(alo, ahi,
          0,1,2,3,4,5,6,7,8,9,10,11,12,13,14,15);
    }
    #pragma unroll
    for (int nt = 0; nt < 2; ++nt) {
      // B fragment: lane holds column rl, K(e) = 16*(e>>3) + 8*hh + (e&7) (matches A)
      const int br = (wn * 32 + nt * 16 + rl) * 32 + 8 * hh;
      v8bf blo = *(const v8bf*)&Bs[br];
      v8bf bhi = *(const v8bf*)&Bs[br + 16];
      v16bf bfrag = __builtin_shufflevector(blo, bhi,
          0,1,2,3,4,5,6,7,8,9,10,11,12,13,14,15);
      #pragma unroll
      for (int mt = 0; mt < 2; ++mt)
        acc[mt][nt] = __builtin_amdgcn_wmma_f32_16x16x32_bf16(
            false, afrag[mt], false, bfrag, (short)0, acc[mt][nt], false, false);
    }
    __syncthreads();
  }

  // ---- epilogue: C fragment element e -> row_local = 8*hh + e, col_local = rl ----
  const int posv = (EPI == 1) ? pos_off[0] : 0;
  const float inv_n = (EPI == 1) ? (1.0f / (float)ldc) : 0.f;
  #pragma unroll
  for (int mt = 0; mt < 2; ++mt) {
    #pragma unroll
    for (int nt = 0; nt < 2; ++nt) {
      #pragma unroll
      for (int e = 0; e < 8; ++e) {
        int row = m0 + wm * 32 + mt * 16 + 8 * hh + e;
        int col = n0 + wn * 32 + nt * 16 + rl;
        float v = acc[mt][nt][e];
        if (EPI == 0) {
          ((float*)Cv)[(long)hz * sC + (long)row * ldc + col] = v;
        } else if (EPI == 1) {
          v = fmaxf(v, 0.f);                                   // relu -> x_sparse
          bf16* C = (bf16*)Cv + (long)hz * sC;
          C[(long)row * ldc + col] = (bf16)v;
          // rope: pairs (2i, 2i+1) live in adjacent lanes (col differs in rl bit0)
          float pv = __shfl_xor(v, 1, 32);
          float vr = (col & 1) ? pv : -pv;
          int q = col & ~1;                                    // quantize(n, 2)
          // freq = THETA^(-q/N)/2pi, THETA = 2^16, N == ldc for this call
          float freq = exp2f(-16.0f * (float)q * inv_n) * 0.15915494309189535f;
          float cyc = (float)(posv + row) * freq;
          float ph = (cyc - floorf(cyc)) * 6.283185307179586f;
          float qv = v * __cosf(ph) + vr * __sinf(ph);
          bf16* O2 = out2 + (long)hz * sC;
          O2[(long)row * ldc + col] = (bf16)qv;
        } else if (EPI == 2) {
          bf16* C = (bf16*)Cv + (long)hz * sC;                 // strictly-lower mask
          C[(long)row * ldc + col] = (bf16)((col < row) ? v : 0.f);
        } else if (EPI == 3) {
          v = fmaxf(v, 0.f);
          float xs = (float)aux[(long)hz * sAux + (long)row * ldaux + col];
          bf16* C = (bf16*)Cv + (long)hz * sC;
          C[(long)row * ldc + col] = (bf16)(v * xs);
        } else {                                               // EPI == 4: NT f32
          __builtin_nontemporal_store(v,
              &((float*)Cv)[(long)hz * sC + (long)row * ldc + col]);
        }
      }
    }
  }
}

// ---------------------------------------------------------------------------
extern "C" void kernel_launch(void* const* d_in, const int* in_sizes, int n_in,
                              void* d_out, int out_size, void* d_ws, size_t ws_size,
                              hipStream_t stream) {
  (void)in_sizes; (void)n_in; (void)out_size; (void)ws_size;
  const int*   idx       = (const int*)d_in[0];
  const float* prev_rho  = (const float*)d_in[1];
  const int*   pos_off   = (const int*)d_in[2];
  const float* embed_w   = (const float*)d_in[3];
  const float* encoder   = (const float*)d_in[4];
  const float* encoder_v = (const float*)d_in[5];
  const float* decoder   = (const float*)d_in[6];
  const float* lm_head   = (const float*)d_in[7];
  float* out = (float*)d_out;

  // ---- workspace carve-out (~231 MB) ----
  char* ws = (char*)d_ws;
  auto carve = [&](size_t bytes) { char* p = ws; ws += (bytes + 255) & ~(size_t)255; return p; };
  bf16*  enc_b  = (bf16*)carve((size_t)NH * D_DIM * N_DIM * 2);
  bf16*  encv_b = (bf16*)carve((size_t)NH * D_DIM * N_DIM * 2);
  bf16*  dec_b  = (bf16*)carve((size_t)HN * D_DIM * 2);
  bf16*  lm_b   = (bf16*)carve((size_t)D_DIM * V_DIM * 2);
  float* x_f    = (float*)carve((size_t)T_SEQ * D_DIM * 4);
  bf16*  x_b    = (bf16*)carve((size_t)T_SEQ * D_DIM * 2);
  bf16*  xs_b   = (bf16*)carve((size_t)NH * T_SEQ * N_DIM * 2);
  bf16*  qr_b   = (bf16*)carve((size_t)NH * T_SEQ * N_DIM * 2);
  bf16*  sc_b   = (bf16*)carve((size_t)NH * T_SEQ * T_SEQ * 2);
  float* ykv_f  = (float*)carve((size_t)NH * T_SEQ * D_DIM * 4);
  bf16*  ykv_b  = (bf16*)carve((size_t)NH * T_SEQ * D_DIM * 2);
  bf16*  xy_b   = (bf16*)carve((size_t)T_SEQ * HN * 2);
  float* ymlp_f = (float*)carve((size_t)T_SEQ * D_DIM * 4);
  float* xm     = (float*)carve((size_t)NLAYER * HN * 4);
  float* vm     = (float*)carve((size_t)NLAYER * D_DIM * 4);

  // ---- weight conversion fp32 -> bf16 ----
  k_f32_to_bf16<<<2048, 256, 0, stream>>>(encoder,   enc_b,  (long)NH * D_DIM * N_DIM);
  k_f32_to_bf16<<<2048, 256, 0, stream>>>(encoder_v, encv_b, (long)NH * D_DIM * N_DIM);
  k_f32_to_bf16<<<2048, 256, 0, stream>>>(decoder,   dec_b,  (long)HN * D_DIM);
  k_f32_to_bf16<<<2048, 256, 0, stream>>>(lm_head,   lm_b,   (long)D_DIM * V_DIM);

  // ---- x = ln(embed_w[idx]) ----
  k_embed_ln<<<T_SEQ, 256, 0, stream>>>(idx, embed_w, x_f, x_b);

  const long sEnc = (long)D_DIM * N_DIM;
  const long sXS  = (long)T_SEQ * N_DIM;
  const long sSC  = (long)T_SEQ * T_SEQ;
  const long sKV  = (long)T_SEQ * D_DIM;

  for (int l = 0; l < NLAYER; ++l) {
    // x_sparse = relu(x @ enc[h]); QR = rope(x_sparse)
    k_gemm<1, false><<<dim3(N_DIM/64, T_SEQ/128, NH), 256, 0, stream>>>(
        x_b, enc_b, xs_b, nullptr, qr_b,
        D_DIM, D_DIM, N_DIM, N_DIM, 0, 0, sEnc, sXS, 0, pos_off);
    // per-layer means for the Hebbian rho update
    k_colmean_bf16<<<dim3(N_DIM/256, NH), 256, 0, stream>>>(
        xs_b, xm + (long)l * HN, T_SEQ, N_DIM, sXS);
    k_colmean_f32<<<D_DIM/256, 256, 0, stream>>>(x_f, vm + l * D_DIM, T_SEQ, D_DIM);
    // scores = tril(QR @ QR^T, -1)
    k_gemm<2, true><<<dim3(T_SEQ/64, T_SEQ/128, NH), 256, 0, stream>>>(
        qr_b, qr_b, sc_b, nullptr, nullptr,
        N_DIM, N_DIM, N_DIM, T_SEQ, 0, sXS, sXS, sSC, 0, nullptr);
    // yKV_raw = scores @ x   (then row-LN -> bf16)
    k_gemm<0, false><<<dim3(D_DIM/64, T_SEQ/128, NH), 256, 0, stream>>>(
        sc_b, x_b, ykv_f, nullptr, nullptr,
        T_SEQ, T_SEQ, D_DIM, D_DIM, 0, sSC, 0, sKV, 0, nullptr);
    k_ln_rows<<<NH * T_SEQ, 256, 0, stream>>>(ykv_f, ykv_b);
    // xy = relu(yKV @ encv[h]) * x_sparse, scattered to [T, H*N]
    k_gemm<3, false><<<dim3(N_DIM/64, T_SEQ/128, NH), 256, 0, stream>>>(
        ykv_b, encv_b, xy_b, xs_b, nullptr,
        D_DIM, D_DIM, N_DIM, HN, N_DIM, sKV, sEnc, (long)N_DIM, sXS, nullptr);
    // yMLP = xy @ decoder  (K = 16384)
    k_gemm<0, false><<<dim3(D_DIM/64, T_SEQ/128, 1), 256, 0, stream>>>(
        xy_b, dec_b, ymlp_f, nullptr, nullptr,
        HN, HN, D_DIM, D_DIM, 0, 0, 0, 0, 0, nullptr);
    // y = ln(yMLP); x = ln(x + y)
    k_ln_residual<<<T_SEQ, 256, 0, stream>>>(ymlp_f, x_f, x_b);
  }

  // logits = x @ lm_head  -> d_out[0 .. T*V), streamed (non-temporal)
  k_gemm<4, false><<<dim3(V_DIM/64, T_SEQ/128, 1), 256, 0, stream>>>(
      x_b, lm_b, out, nullptr, nullptr,
      D_DIM, D_DIM, V_DIM, V_DIM, 0, 0, 0, 0, 0, nullptr);

  // new_rho and total_rho_update after the logits block
  float* out_new = out + (long)T_SEQ * V_DIM;
  float* out_upd = out_new + (long)HND;
  k_rho_final<<<HND / 256, 256, 0, stream>>>(xm, vm, prev_rho, out_new, out_upd);
}
